// EquationSampler_20486994002083
// MI455X (gfx1250) — compile-verified, matrix operand order checked by reading, not run on hardware
//
#include <hip/hip_runtime.h>
#include <hip/hip_bf16.h>
#include <math.h>

// ---------------------------------------------------------------------------
// EquationSampler on MI455X (gfx1250): 12-step autoregressive RNN sampler.
//
// Mapping: 256 threads = 8 wave32s per block; EACH WAVE owns an independent
// 16-sample WMMA M-tile (128 samples/block, 512 blocks for n=65536).
// Shared (read-only after init): bf16-transposed Wx/Wh/Wp + f32 biases.
// Per-wave private LDS slices: h (16x128 bf16), x (16x32 bf16), logits (16x16
// f32).  Since a wave's LDS ops complete in order (DScnt ordering) and each
// wave only touches its own slices, NO workgroup barriers are needed inside
// the 12-step loop — only one after the weight preload.
//
// Per wave per step (44 v_wmma_f32_16x16x32_bf16, zero idle waves):
//   regs:   ax = A-frag(x), ah[0..3] = A-frags(h)
//   nt=0..7: acc = ax@WxT[nt] + sum_kt ah[kt]@WhT[nt,kt]; tanh(+b) -> sH
//   logits: 4 WMMAs of new-h @ WpT -> sLogits
//   lanes 0..15: softmax + grammar constraints + categorical sample +
//                parent/sibling scan + one-hot x rebuild (scalar, f32)
// tanh uses the gfx1250 hardware V_TANH_F32 transcendental when the builtin
// exists; otherwise a branchless exp-based formula (no exec-mask branches).
// ---------------------------------------------------------------------------

typedef __attribute__((ext_vector_type(16))) __bf16 v16bf;
typedef __attribute__((ext_vector_type(8)))  float  v8f;

#define N_TOKENS   16
#define MAX_STEPS  12
#define HIDDEN     128
#define IN_SIZE    32
#define WAVES      8
#define M_PER_BLK  (WAVES * 16)

__device__ __forceinline__ __bf16 f2bf(float f) {
    union { float f; unsigned u; } c; c.f = f;
    unsigned r = c.u + 0x7FFFu + ((c.u >> 16) & 1u);   // round-to-nearest-even
    unsigned short h = (unsigned short)(r >> 16);
    __bf16 o;
    __builtin_memcpy(&o, &h, 2);
    return o;
}

// Hardware tanh (V_TANH_F32) if available, else branchless exp-based tanh.
__device__ __forceinline__ float fast_tanh(float x) {
#if defined(__has_builtin)
#if __has_builtin(__builtin_amdgcn_tanhf)
    return __builtin_amdgcn_tanhf(x);
#else
    float ax = fabsf(x);
    float e  = __expf(-2.0f * ax);          // one v_exp_f32
    float t  = (1.0f - e) / (1.0f + e);     // tanh(|x|), branchless
    return copysignf(t, x);
#endif
#else
    float ax = fabsf(x);
    float e  = __expf(-2.0f * ax);
    float t  = (1.0f - e) / (1.0f + e);
    return copysignf(t, x);
#endif
}

// Deterministic counter-based uniform in [0,1): splitmix64 finalizer.
__device__ __forceinline__ float rng01(unsigned long long x) {
    x += 0x9E3779B97F4A7C15ull;
    x ^= x >> 33; x *= 0xFF51AFD7ED558CCDull;
    x ^= x >> 33; x *= 0xC4CEB9FE1A85EC53ull;
    x ^= x >> 33;
    return (float)(unsigned)(x >> 40) * (1.0f / 16777216.0f);
}

// Load a 16x32 bf16 A/B fragment from LDS (row-major, K contiguous, stride ld).
// CDNA5 16-bit A layout: lanes 0-15 row=lane (K base +0), lanes 16-31
// row=lane-16 (K base +8); VGPRs 0-3 = K 0..15 region, VGPRs 4-7 = K 16..31.
__device__ __forceinline__ v16bf load_frag(const __bf16* base, int ld, int koff) {
    const int lane = threadIdx.x & 31;
    const int row  = lane & 15;
    const int g    = lane >> 4;
    const __bf16* p = base + row * ld + koff + 8 * g;
    v16bf f;
#pragma unroll
    for (int v = 0; v < 4; ++v) {
        f[2 * v]         = p[2 * v];
        f[2 * v + 1]     = p[2 * v + 1];
        f[8 + 2 * v]     = p[16 + 2 * v];
        f[8 + 2 * v + 1] = p[16 + 2 * v + 1];
    }
    return f;
}

__device__ __forceinline__ v8f wmma_bf16(v16bf a, v16bf b, v8f c) {
    return __builtin_amdgcn_wmma_f32_16x16x32_bf16(
        /*neg_a=*/false, a, /*neg_b=*/false, b,
        /*c_mod=*/(short)0, c, /*reuse_a=*/false, /*reuse_b=*/false);
}

// dynamic-LDS layout (bytes), 16B aligned sections
#define OFF_WX   0                                   // 128*32 bf16  =  8192 B
#define OFF_WH   (OFF_WX + HIDDEN * IN_SIZE * 2)     // 128*128 bf16 = 32768 B
#define OFF_WP   (OFF_WH + HIDDEN * HIDDEN * 2)      // 16*128 bf16  =  4096 B
#define OFF_B    (OFF_WP + N_TOKENS * HIDDEN * 2)    // 128 f32      =   512 B
#define OFF_BP   (OFF_B + HIDDEN * 4)                // 16 f32       =    64 B
#define OFF_X    (OFF_BP + N_TOKENS * 4)             // 8*16*32 bf16 =  8192 B
#define OFF_H    (OFF_X + WAVES * 16 * IN_SIZE * 2)  // 8*16*128 bf16= 32768 B
#define OFF_LG   (OFF_H + WAVES * 16 * HIDDEN * 2)   // 8*16*16 f32  =  8192 B
#define SMEM_BYTES (OFF_LG + WAVES * 16 * N_TOKENS * 4)

__global__ __launch_bounds__(256)
void EquationSampler_kernel(const float* __restrict__ x0,
                            const float* __restrict__ h0,
                            const float* __restrict__ Wx,   // [32][128]
                            const float* __restrict__ Wh,   // [128][128]
                            const float* __restrict__ b,    // [128]
                            const float* __restrict__ Wp,   // [128][16]
                            const float* __restrict__ bp,   // [16]
                            float* __restrict__ out, int n) {
    extern __shared__ __align__(16) char smem[];
    __bf16* sWx = (__bf16*)(smem + OFF_WX);
    __bf16* sWh = (__bf16*)(smem + OFF_WH);
    __bf16* sWp = (__bf16*)(smem + OFF_WP);
    float*  sB  = (float*)(smem + OFF_B);
    float*  sBp = (float*)(smem + OFF_BP);

    const int tid  = threadIdx.x;
    const int wave = tid >> 5;
    const int lane = tid & 31;

    __bf16* sX  = (__bf16*)(smem + OFF_X) + wave * 16 * IN_SIZE;   // [m][k]
    __bf16* sH  = (__bf16*)(smem + OFF_H) + wave * 16 * HIDDEN;    // [m][k]
    float*  sLg = (float*)(smem + OFF_LG) + wave * 16 * N_TOKENS;  // [m][tok]

    const int gid = blockIdx.x * M_PER_BLK + wave * 16 + (lane & 15);

    // ---- one-time weight conversion f32 -> bf16, transposed into LDS ----
    for (int i = tid; i < HIDDEN * IN_SIZE; i += 256) {
        int nn = i >> 5, k = i & 31;
        sWx[nn * IN_SIZE + k] = f2bf(Wx[k * HIDDEN + nn]);
    }
    for (int i = tid; i < HIDDEN * HIDDEN; i += 256) {
        int nn = i >> 7, k = i & 127;
        sWh[nn * HIDDEN + k] = f2bf(Wh[k * HIDDEN + nn]);
    }
    for (int i = tid; i < N_TOKENS * HIDDEN; i += 256) {
        int tk = i >> 7, k = i & 127;
        sWp[tk * HIDDEN + k] = f2bf(Wp[k * N_TOKENS + tk]);
    }
    for (int i = tid; i < HIDDEN; i += 256) sB[i] = b[i];
    if (tid < N_TOKENS) sBp[tid] = bp[tid];
    // initial per-wave x / h (broadcast row) — each wave fills its own slice
    for (int i = lane; i < 16 * IN_SIZE; i += 32) sX[i] = f2bf(x0[i & 31]);
    for (int i = lane; i < 16 * HIDDEN;  i += 32) sH[i] = f2bf(h0[i & 127]);

    // ---- output segment pointers (seq, ent, lp, cnt, len, mask) ----
    float* seqO  = out;
    float* entO  = out + (size_t)n * 12;
    float* lpO   = out + (size_t)n * 24;
    float* cntO  = out + (size_t)n * 36;
    float* lenO  = cntO + n;
    float* maskO = lenO + n;

    // per-sample state (lives in the sampling lane, lane < 16)
    int   lseq[MAX_STEPS];
    int   lcnt = 1, lln = 0, lact = 1, lhasvar = 0, llen = 1;
    if (lane < 16) {
#pragma unroll
        for (int i = 0; i < MAX_STEPS; ++i) lseq[i] = -1;
        if (gid < n) maskO[(size_t)gid * 13] = 1.0f;
    }
    __syncthreads();   // weights visible to all waves; only barrier needed

    for (int t = 0; t < MAX_STEPS; ++t) {
        // ---- A fragments for this step (registers; old h/x snapshot) ----
        v16bf ax  = load_frag(sX, IN_SIZE, 0);
        v16bf ah0 = load_frag(sH, HIDDEN, 0);
        v16bf ah1 = load_frag(sH, HIDDEN, 32);
        v16bf ah2 = load_frag(sH, HIDDEN, 64);
        v16bf ah3 = load_frag(sH, HIDDEN, 96);

        // ---- hidden update, 8 N-tiles; write tanh'd bf16 back to sH ----
        const int g    = lane >> 4;
        const int ncol = lane & 15;
#pragma unroll
        for (int nt = 0; nt < 8; ++nt) {
            v8f acc = {};
            acc = wmma_bf16(ax,  load_frag(sWx + (nt << 4) * IN_SIZE, IN_SIZE, 0),  acc);
            acc = wmma_bf16(ah0, load_frag(sWh + (nt << 4) * HIDDEN, HIDDEN, 0),   acc);
            acc = wmma_bf16(ah1, load_frag(sWh + (nt << 4) * HIDDEN, HIDDEN, 32),  acc);
            acc = wmma_bf16(ah2, load_frag(sWh + (nt << 4) * HIDDEN, HIDDEN, 64),  acc);
            acc = wmma_bf16(ah3, load_frag(sWh + (nt << 4) * HIDDEN, HIDDEN, 96),  acc);
            int   col = (nt << 4) + ncol;
            float bb  = sB[col];
#pragma unroll
            for (int v = 0; v < 8; ++v) {
                int mrow = v + 8 * g;
                sH[mrow * HIDDEN + col] = f2bf(fast_tanh(acc[v] + bb));
            }
        }
        asm volatile("" ::: "memory");  // order sH writes before re-reads (LDS is in-order per wave)

        // ---- logits = h' @ Wp (this wave's own tile) ----
        {
            v8f lacc = {};
            v16bf nh0 = load_frag(sH, HIDDEN, 0);
            v16bf nh1 = load_frag(sH, HIDDEN, 32);
            v16bf nh2 = load_frag(sH, HIDDEN, 64);
            v16bf nh3 = load_frag(sH, HIDDEN, 96);
            lacc = wmma_bf16(nh0, load_frag(sWp, HIDDEN, 0),  lacc);
            lacc = wmma_bf16(nh1, load_frag(sWp, HIDDEN, 32), lacc);
            lacc = wmma_bf16(nh2, load_frag(sWp, HIDDEN, 64), lacc);
            lacc = wmma_bf16(nh3, load_frag(sWp, HIDDEN, 96), lacc);
#pragma unroll
            for (int v = 0; v < 8; ++v)
                sLg[(v + 8 * g) * N_TOKENS + ncol] = lacc[v];
        }
        asm volatile("" ::: "memory");

        // ---- per-sample sampling (lanes 0..15 of each wave) ----
        if (lane < 16 && gid < n) {
            const int m = lane;
            float p[N_TOKENS];
            float mx = -1e30f;
#pragma unroll
            for (int i = 0; i < N_TOKENS; ++i) {
                p[i] = sLg[m * N_TOKENS + i] + sBp[i];
                mx = fmaxf(mx, p[i]);
            }
            float sum = 0.f;
#pragma unroll
            for (int i = 0; i < N_TOKENS; ++i) { p[i] = __expf(p[i] - mx); sum += p[i]; }
            float inv = 1.f / sum;
#pragma unroll
            for (int i = 0; i < N_TOKENS; ++i) p[i] *= inv;

            // grammar constraints (pre-update cnt/ln/seq)
            int cl = lcnt + lln;
            if (cl < 2) {                 // min length: kill zero-arity tokens
#pragma unroll
                for (int i = 8; i < 16; ++i) p[i] = 0.f;
            }
            if (cl > MAX_STEPS - 2) {     // max length: kill expanding tokens
#pragma unroll
                for (int i = 0; i < 8; ++i) p[i] = 0.f;
            }
            if (lcnt == 1 && !lhasvar) p[8] = 0.f;  // last slot must allow a var

            float s2 = 0.f;
#pragma unroll
            for (int i = 0; i < N_TOKENS; ++i) s2 += p[i];
            float u = rng01(((unsigned long long)gid * MAX_STEPS + t) ^ 0xD1B54A32D192ED03ull) * s2;

            int tok = -1, lastpos = 0;
            float csum = 0.f;
#pragma unroll
            for (int i = 0; i < N_TOKENS; ++i) {
                if (p[i] > 0.f) {
                    lastpos = i;
                    csum += p[i];
                    if (tok < 0 && u < csum) tok = i;
                }
            }
            if (tok < 0) tok = lastpos;

            float is2 = 1.f / s2;
            float lp  = __logf(p[tok] * is2);
            float ent = 0.f;
#pragma unroll
            for (int i = 0; i < N_TOKENS; ++i) {
                if (p[i] > 0.f) { float pn = p[i] * is2; ent -= pn * __logf(pn); }
            }

            seqO[(size_t)gid * 12 + t] = (float)tok;
            entO[(size_t)gid * 12 + t] = ent;
            lpO [(size_t)gid * 12 + t] = lp;

            // state update
            int ar = (tok < 4) ? 2 : ((tok < 8) ? 1 : 0);
            lcnt = lcnt - 1 + ar;
            lln += 1;
            lact = (lcnt > 0) && lact;
            lseq[t] = tok;
            if (tok >= 9) lhasvar = 1;
            if (lact) llen += 1;
            maskO[(size_t)gid * 13 + (t + 1)] = (float)lact;

            // parent / sibling scan (back over emitted tokens)
            int ps0 = -1, ps1 = -1, found = 0;
            float c2 = 0.f;
            for (int i = t; i >= 0; --i) {
                int tk2 = lseq[i];
                float arity = (tk2 >= 0 && tk2 < 4) ? 2.f : ((tk2 >= 4 && tk2 < 8) ? 1.f : 0.f);
                c2 += arity - 1.f;
                if (!found && c2 == 0.f) {
                    ps0 = tk2;
                    ps1 = (i + 1 < MAX_STEPS) ? lseq[i + 1] : -1;
                    found = 1;
                }
            }
            if (tok < 8) { ps0 = tok; ps1 = -1; }   // most-recent is an operator

            // rebuild one-hot x for next step
#pragma unroll
            for (int k = 0; k < IN_SIZE; ++k) sX[m * IN_SIZE + k] = f2bf(0.f);
            if (ps0 >= 0 && ps0 < N_TOKENS) sX[m * IN_SIZE + ps0] = f2bf(1.f);
            if (ps1 >= 0 && ps1 < N_TOKENS) sX[m * IN_SIZE + N_TOKENS + ps1] = f2bf(1.f);
        }
        asm volatile("" ::: "memory");
    }

    if (lane < 16 && gid < n) {
        cntO[gid] = (float)lcnt;
        lenO[gid] = (float)llen;
    }
}

extern "C" void kernel_launch(void* const* d_in, const int* in_sizes, int n_in,
                              void* d_out, int out_size, void* d_ws, size_t ws_size,
                              hipStream_t stream) {
    (void)in_sizes; (void)n_in; (void)d_ws; (void)ws_size;
    // outputs per sample: 12 seq + 12 ent + 12 lp + 1 cnt + 1 len + 13 mask = 51
    const int n = out_size / 51;
    const float* x0 = (const float*)d_in[1];
    const float* h0 = (const float*)d_in[2];
    const float* Wx = (const float*)d_in[3];
    const float* Wh = (const float*)d_in[4];
    const float* bb = (const float*)d_in[5];
    const float* Wp = (const float*)d_in[6];
    const float* bp = (const float*)d_in[7];
    float* out = (float*)d_out;

    const int blocks = (n + M_PER_BLK - 1) / M_PER_BLK;
    EquationSampler_kernel<<<blocks, 256, SMEM_BYTES, stream>>>(x0, h0, Wx, Wh, bb, Wp, bp, out, n);
}
